// SynthesisLayer_13932873909235
// MI455X (gfx1250) — matrix operand order repaired and checked by reading, not compiled, not run on hardware
//
#include <hip/hip_runtime.h>

typedef _Float16 v8h  __attribute__((ext_vector_type(8)));
typedef _Float16 v16h __attribute__((ext_vector_type(16)));
typedef float    v8f  __attribute__((ext_vector_type(8)));

#define BATCH 8
#define CIN   512
#define COUT  512
#define RES   64
#define WDIM  512

#define CH     64          // input channels staged per LDS chunk
#define NCOL   66          // columns -1..64 (zero-padded borders for kw shift)
#define CHP    72          // padded channel stride in LDS (f16 units) -> bank-safe
#define MTILES 2           // 32 COUT rows per wave
#define NTILES 4           // 64 pixels per wave
#define NTAPS  9
#define AW_ELEMS (NTAPS * COUT * CIN)   // 2,359,296 f16 per array

// ---------------------------------------------------------------------------
// style[b][c] = affine_b[c] + sum_j w[b][j] * affine_w[c][j]   (8 x 512)
// ---------------------------------------------------------------------------
__global__ __launch_bounds__(256)
void style_kernel(const float* __restrict__ wvec,
                  const float* __restrict__ aw,
                  const float* __restrict__ ab,
                  float* __restrict__ style)
{
    int i = blockIdx.x * 256 + threadIdx.x;      // 0..4095
    int b = i >> 9;
    int c = i & 511;
    const float* wr = wvec + b * WDIM;
    const float* ar = aw   + c * WDIM;
    float acc = ab[c];
    for (int j = 0; j < WDIM; j += 4) {
        acc += wr[j+0]*ar[j+0] + wr[j+1]*ar[j+1]
             + wr[j+2]*ar[j+2] + wr[j+3]*ar[j+3];
    }
    style[i] = acc;
}

// ---------------------------------------------------------------------------
// Split fp32 weight -> (hi,lo) f16 pair, repacked to [tap][o][ch] so each
// WMMA A-fragment is two contiguous b128 loads per lane.
// ---------------------------------------------------------------------------
__global__ __launch_bounds__(256)
void repack_weight(const float* __restrict__ weight,
                   _Float16* __restrict__ ahi,
                   _Float16* __restrict__ alo)
{
    int d   = blockIdx.x * 256 + threadIdx.x;    // [tap][o][ch]
    int ch  = d & 511;
    int t   = d >> 9;
    int o   = t & 511;
    int tap = t >> 9;                            // 0..8
    float f = weight[((size_t)o * CIN + ch) * NTAPS + tap];
    _Float16 hi = (_Float16)f;
    ahi[d] = hi;
    alo[d] = (_Float16)(f - (float)hi);
}

// ---------------------------------------------------------------------------
// Modulated 3x3 conv as 9 shifted GEMMs, split-f16 WMMA (3-term, fp32 acc).
// Grid: (COUT/256, RES, BATCH). Block: 256 threads = 8 waves.
// Block computes out[b, oBlock..oBlock+255, h, 0..63].
// ---------------------------------------------------------------------------
__global__ __launch_bounds__(256)
void modconv_wmma_f16(const float* __restrict__ x,
                      const _Float16* __restrict__ ahi,
                      const _Float16* __restrict__ alo,
                      const float* __restrict__ bias,
                      const float* __restrict__ style,
                      float* __restrict__ out)
{
    __shared__ _Float16 ldsHi[3 * NCOL * CHP];   // 28,512 B
    __shared__ _Float16 ldsLo[3 * NCOL * CHP];   // 28,512 B
    __shared__ float    ldsStyle[CH];

    const int b      = blockIdx.z;
    const int h      = blockIdx.y;
    const int oBlock = blockIdx.x * 256;
    const int tid    = threadIdx.x;
    const int wave   = tid >> 5;
    const int lane   = tid & 31;
    const int laneLo = lane & 15;
    const int laneHi = lane >> 4;
    const int oWave  = oBlock + wave * 32;

    // ---- per-thread staging constants (one div/mod total) ----
    const int  sn      = tid % NCOL;             // LDS column 0..65 (n' = sn-1)
    const int  sr      = tid / NCOL;             // row 0..2 (kh)
    const bool sActive = tid < 3 * NCOL;
    const int  ihr     = h + sr - 1;
    const int  iwr     = sn - 1;
    const bool sPred   = sActive && ((unsigned)ihr < RES) && ((unsigned)iwr < RES);
    const int  ih0     = ((unsigned)ihr < RES) ? ihr : 0;
    const int  iw0     = ((unsigned)iwr < RES) ? iwr : 0;
    const float* xp    = x + (((size_t)(b * CIN) * RES + ih0) * RES + iw0);
    const int  sAddr   = (sr * NCOL + sn) * CHP;

    // ---- per-lane fragment base addresses (f16-element units) ----
    int aBase[MTILES];
#pragma unroll
    for (int mt = 0; mt < MTILES; ++mt)
        aBase[mt] = (oWave + mt * 16 + laneLo) * CIN + laneHi * 8;
    int bBase[NTILES];
#pragma unroll
    for (int nt = 0; nt < NTILES; ++nt)
        bBase[nt] = (nt * 16 + laneLo) * CHP + laneHi * 8;

    v8f acc[MTILES][NTILES] = {};

    for (int cbase = 0; cbase < CIN; cbase += CH) {
        // styles for this chunk (safe: prior compute reads only ldsHi/ldsLo)
        if (tid < CH) ldsStyle[tid] = style[b * CIN + cbase + tid];
        __syncthreads();                          // styles visible, LDS free

        // ---- stage modulated raw rows, split hi/lo f16 ----
        if (sActive) {
#pragma unroll 4
            for (int ch = 0; ch < CH; ++ch) {
                float v = sPred ? xp[(size_t)(cbase + ch) << 12] : 0.0f;
                v *= ldsStyle[ch];
                _Float16 hi = (_Float16)v;
                _Float16 lo = (_Float16)(v - (float)hi);
                ldsHi[sAddr + ch] = hi;
                ldsLo[sAddr + ch] = lo;
            }
        }
        __syncthreads();

        // ---- 9 shifted GEMM taps over this chunk's CH=64 channels ----
#pragma unroll 1
        for (int tap = 0; tap < NTAPS; ++tap) {
            const int kh   = tap / 3;
            const int kw   = tap - 3 * kh;
            const int tapB = (kh * NCOL + kw) * CHP;          // scalar
            const int tapA = tap * (COUT * CIN) + cbase;      // scalar

#pragma unroll
            for (int ks = 0; ks < CH; ks += 32) {
                union { v16h v; v8h h[2]; } Ah[MTILES], Al[MTILES],
                                            Bh[NTILES], Bl[NTILES];
#pragma unroll
                for (int mt = 0; mt < MTILES; ++mt) {
                    int ia = aBase[mt] + tapA + ks;
                    Ah[mt].h[0] = *(const v8h*)(ahi + ia);
                    Ah[mt].h[1] = *(const v8h*)(ahi + ia + 16);
                    Al[mt].h[0] = *(const v8h*)(alo + ia);
                    Al[mt].h[1] = *(const v8h*)(alo + ia + 16);
                }
#pragma unroll
                for (int nt = 0; nt < NTILES; ++nt) {
                    int ib = bBase[nt] + tapB + ks;
                    Bh[nt].h[0] = *(const v8h*)(ldsHi + ib);
                    Bh[nt].h[1] = *(const v8h*)(ldsHi + ib + 16);
                    Bl[nt].h[0] = *(const v8h*)(ldsLo + ib);
                    Bl[nt].h[1] = *(const v8h*)(ldsLo + ib + 16);
                }
                // 3-term split product; same-acc reuse distance = 8 WMMAs
#pragma unroll
                for (int mt = 0; mt < MTILES; ++mt)
#pragma unroll
                    for (int nt = 0; nt < NTILES; ++nt)
                        acc[mt][nt] = __builtin_amdgcn_wmma_f32_16x16x32_f16(
                            false, Ah[mt].v, false, Bh[nt].v,
                            (short)0, acc[mt][nt], false, false);
#pragma unroll
                for (int mt = 0; mt < MTILES; ++mt)
#pragma unroll
                    for (int nt = 0; nt < NTILES; ++nt)
                        acc[mt][nt] = __builtin_amdgcn_wmma_f32_16x16x32_f16(
                            false, Ah[mt].v, false, Bl[nt].v,
                            (short)0, acc[mt][nt], false, false);
#pragma unroll
                for (int mt = 0; mt < MTILES; ++mt)
#pragma unroll
                    for (int nt = 0; nt < NTILES; ++nt)
                        acc[mt][nt] = __builtin_amdgcn_wmma_f32_16x16x32_f16(
                            false, Al[mt].v, false, Bh[nt].v,
                            (short)0, acc[mt][nt], false, false);
            }
        }
    }

    // ---- epilogue: C-matrix layout -> out[b, o, h, w] + bias[o] ----
#pragma unroll
    for (int mt = 0; mt < MTILES; ++mt) {
#pragma unroll
        for (int r = 0; r < 8; ++r) {
            int o    = oWave + mt * 16 + r + 8 * laneHi;
            float bv = bias[o];
#pragma unroll
            for (int nt = 0; nt < NTILES; ++nt) {
                int wcol = nt * 16 + laneLo;
                out[(size_t)((b * COUT + o) * RES + h) * RES + wcol] =
                    acc[mt][nt][r] + bv;
            }
        }
    }
}

// ---------------------------------------------------------------------------
extern "C" void kernel_launch(void* const* d_in, const int* in_sizes, int n_in,
                              void* d_out, int out_size, void* d_ws, size_t ws_size,
                              hipStream_t stream)
{
    const float* x        = (const float*)d_in[0];  // [8,512,64,64]
    const float* w        = (const float*)d_in[1];  // [8,512]
    const float* weight   = (const float*)d_in[2];  // [512,512,3,3]
    const float* bias     = (const float*)d_in[3];  // [512]
    const float* affine_w = (const float*)d_in[4];  // [512,512]
    const float* affine_b = (const float*)d_in[5];  // [512]
    float* out = (float*)d_out;                     // [8,512,64,64]

    // workspace layout: style (16 KB) | Ahi (4.5 MB f16) | Alo (4.5 MB f16)
    float*    style = (float*)d_ws;
    _Float16* ahi   = (_Float16*)((char*)d_ws + 16384);
    _Float16* alo   = ahi + AW_ELEMS;

    style_kernel<<<(BATCH * CIN) / 256, 256, 0, stream>>>(w, affine_w, affine_b, style);
    repack_weight<<<AW_ELEMS / 256, 256, 0, stream>>>(weight, ahi, alo);

    dim3 grid(COUT / 256, RES, BATCH);              // (2, 64, 8)
    modconv_wmma_f16<<<grid, 256, 0, stream>>>(x, ahi, alo, bias, style, out);
}
